// BertSelfAttention_8169027797086
// MI455X (gfx1250) — matrix operand order repaired and checked by reading, not compiled, not run on hardware
//
#include <hip/hip_runtime.h>
#include <hip/hip_bf16.h>
#include <stdint.h>

#define BB 4
#define SS 2048
#define DD 1024
#define HH 16
#define HD 64

typedef __bf16 bf16;
typedef __attribute__((ext_vector_type(16))) __bf16 v16bf;
typedef __attribute__((ext_vector_type(8)))  float  v8f;
typedef __attribute__((__vector_size__(16))) int v4i;

union Frag {
  uint4 q[2];
  v16bf v;
};

static __device__ __forceinline__ v8f wmma_bf16(v16bf a, v16bf b, v8f c) {
  // 8 args: (neg_a, A, neg_b, B, c_mod, C, reuse_a, reuse_b)
  return __builtin_amdgcn_wmma_f32_16x16x32_bf16(false, a, false, b, (short)0, c,
                                                 false, false);
}

static __device__ __forceinline__ v8f zero8() {
  v8f z;
#pragma unroll
  for (int i = 0; i < 8; ++i) z[i] = 0.0f;
  return z;
}

// ---- CDNA5 async global->LDS copy (ASYNCcnt path) -------------------------
static __device__ __forceinline__ void async_copy_b128(const bf16* g, bf16* l) {
#if __has_builtin(__builtin_amdgcn_global_load_async_to_lds_b128)
  const v4i* gg = (const v4i*)g;
  v4i* ll = (v4i*)l;
  __builtin_amdgcn_global_load_async_to_lds_b128(
      (__attribute__((address_space(1))) v4i*)gg,
      (__attribute__((address_space(3))) v4i*)ll, 0, 0);
#else
  unsigned loff = (unsigned)(uintptr_t)(__attribute__((address_space(3))) bf16*)l;
  asm volatile("global_load_async_to_lds_b128 %0, %1, off" ::"v"(loff),
               "v"((unsigned long long)(uintptr_t)g)
               : "memory");
#endif
}

template <int N>
static __device__ __forceinline__ void wait_asynccnt() {
#if __has_builtin(__builtin_amdgcn_s_wait_asynccnt)
  __builtin_amdgcn_s_wait_asynccnt((unsigned short)N);
#else
  if constexpr (N == 0)
    asm volatile("s_wait_asynccnt 0x0" ::: "memory");
  else if constexpr (N == 2)
    asm volatile("s_wait_asynccnt 0x2" ::: "memory");
  else
    asm volatile("s_wait_asynccnt 0x4" ::: "memory");
#endif
}

// ---------------------------------------------------------------------------
// fp32 -> bf16 elementwise convert
// ---------------------------------------------------------------------------
__global__ __launch_bounds__(256) void f32_to_bf16_kernel(
    const float* __restrict__ in, bf16* __restrict__ out, int n) {
  int i = blockIdx.x * blockDim.x + threadIdx.x;
  if (i < n) out[i] = (bf16)in[i];
}

// ---------------------------------------------------------------------------
// W [E, K] fp32 -> Wt [K, E] bf16 (16x16 LDS tile transpose)
// ---------------------------------------------------------------------------
__global__ __launch_bounds__(256) void transpose_w_kernel(
    const float* __restrict__ W, bf16* __restrict__ Wt) {
  __shared__ float tile[16][17];
  int n = blockIdx.y * 16 + threadIdx.y;  // row of W (output feature e)
  int k = blockIdx.x * 16 + threadIdx.x;  // col of W (input feature d)
  tile[threadIdx.y][threadIdx.x] = W[n * DD + k];
  __syncthreads();
  int ko = blockIdx.x * 16 + threadIdx.y;
  int no = blockIdx.y * 16 + threadIdx.x;
  Wt[ko * DD + no] = (bf16)tile[threadIdx.x][threadIdx.y];
}

// ---------------------------------------------------------------------------
// Y = X * Wt + bias, scaled, scattered to per-head layout.
//   X  : [8192, 1024] bf16 row-major;  Wt : [1024, 1024] bf16 (= W^T)
// mode 0: out[((b*H+h)*S + s)*64 + hd]   (Q, V layout)
// mode 1: out[((b*H+h)*64 + hd)*S + s]   (K^T layout)
// Block: 128 thr = 4 waves, tile 128x128, each wave 64x64 (4x4 WMMA tiles)
// ---------------------------------------------------------------------------
__global__ __launch_bounds__(128) void qkv_gemm_kernel(
    const bf16* __restrict__ X, const bf16* __restrict__ Wt,
    const float* __restrict__ bias, bf16* __restrict__ out,
    float scale, int mode) {
  const int K = DD, N = DD;
  int wid = threadIdx.x >> 5;
  int lane = threadIdx.x & 31;
  int wm = wid >> 1;  // 0..1
  int wn = wid & 1;   // 0..1
  int m0 = blockIdx.y * 128 + wm * 64;
  int n0 = blockIdx.x * 128 + wn * 64;
  int lrow = lane & 15;
  int lhi = lane >> 4;

  v8f acc[4][4];
#pragma unroll
  for (int i = 0; i < 4; ++i)
#pragma unroll
    for (int j = 0; j < 4; ++j) acc[i][j] = zero8();

  for (int kb = 0; kb < K; kb += 32) {
    // A fragments (16x32): lane<16 -> K chunks [0..7],[16..23];
    // lane>=16 -> [8..15],[24..31]. Two contiguous 16B loads.
    Frag a[4];
#pragma unroll
    for (int i = 0; i < 4; ++i) {
      const bf16* p = X + (size_t)(m0 + i * 16 + lrow) * K + kb + lhi * 8;
      a[i].q[0] = *(const uint4*)p;
      a[i].q[1] = *(const uint4*)(p + 16);
    }
    // B fragments (32x16): lane l holds N=l&15; elem s holds K=(s&7)+8*(l>>4)+16*(s>>3).
    Frag bfr[4];
#pragma unroll
    for (int j = 0; j < 4; ++j) {
      const bf16* p = Wt + (size_t)kb * N + n0 + j * 16 + lrow;
#pragma unroll
      for (int s = 0; s < 16; ++s)
        bfr[j].v[s] = p[(size_t)((s & 7) + lhi * 8 + (s >> 3) * 16) * N];
    }
#pragma unroll
    for (int i = 0; i < 4; ++i)
#pragma unroll
      for (int j = 0; j < 4; ++j)
        acc[i][j] = wmma_bf16(a[i].v, bfr[j].v, acc[i][j]);
  }

  // Epilogue: C layout elem (VGPR r, lane l) = (M = r + 8*(l>>4), N = l&15)
#pragma unroll
  for (int i = 0; i < 4; ++i) {
#pragma unroll
    for (int j = 0; j < 4; ++j) {
#pragma unroll
      for (int r = 0; r < 8; ++r) {
        int m = m0 + i * 16 + r + 8 * lhi;
        int n = n0 + j * 16 + lrow;
        float v = (acc[i][j][r] + bias[n]) * scale;
        int bb = m >> 11;  // batch
        int s = m & 2047;  // seq pos
        int hh = n >> 6;   // head
        int hd = n & 63;   // head dim
        size_t o = (mode == 0)
                       ? (((size_t)(bb * HH + hh) * SS + s) * HD + hd)
                       : (((size_t)(bb * HH + hh) * HD + hd) * SS + s);
        out[o] = (bf16)v;
      }
    }
  }
}

// ---------------------------------------------------------------------------
// Flash attention, 64 keys per iteration. One wave owns 16 queries; the 8
// waves of a block share (b,h) and stream identical key blocks, so K/V are
// staged in LDS with async-to-LDS copies, double-buffered on ASYNCcnt.
//   sK [64][64] : Kt tile  (row d, contiguous keys  -> QK^T B layout)
//   sV [64][64] : V tile   (row key, contiguous hd  -> PV  B layout)
// Per iteration: 8 WMMA (QK^T) + 8 WMMA (PV) vs one softmax reduction pass.
// ---------------------------------------------------------------------------
__global__ __launch_bounds__(256) void attention_kernel(
    const bf16* __restrict__ Q, const bf16* __restrict__ Kt,
    const bf16* __restrict__ V, const float* __restrict__ mask,
    float* __restrict__ out) {
  int wid = threadIdx.x >> 5;
  int lane = threadIdx.x & 31;
  int lrow = lane & 15;
  int lhi = lane >> 4;

  int bh = blockIdx.x >> 4;                     // 16 blocks per (b,h)
  int q0 = (blockIdx.x & 15) * 128 + wid * 16;  // this wave's query tile
  int b = bh >> 4;
  int h = bh & 15;

  __shared__ alignas(16) bf16 sK[2][64 * 64];
  __shared__ alignas(16) bf16 sV[2][64 * 64];
  __shared__ alignas(16) bf16 plds[8][16 * 64];

  // staging split: 512 b128 chunks per tile, 2 per thread
  int t = threadIdx.x;
  int c0 = t * 2;
  int kr0 = c0 >> 3, ko0 = (c0 & 7) * 8;      // chunk c0: row, elem offset
  int kr1 = (c0 + 1) >> 3, ko1 = ((c0 + 1) & 7) * 8;
  const bf16* gK0 = Kt + ((size_t)bh * HD + kr0) * SS + ko0;  // + kb
  const bf16* gK1 = Kt + ((size_t)bh * HD + kr1) * SS + ko1;  // + kb
  const bf16* gV0 = V + ((size_t)bh * SS + kr0) * HD + ko0;   // + kb*HD
  const bf16* gV1 = V + ((size_t)bh * SS + kr1) * HD + ko1;   // + kb*HD

  // Q A-fragments over head dim (two 16x32 tiles cover HD=64)
  Frag qf[2];
#pragma unroll
  for (int i = 0; i < 2; ++i) {
    const bf16* p = Q + ((size_t)bh * SS + q0 + lrow) * HD + i * 32 + lhi * 8;
    qf[i].q[0] = *(const uint4*)p;
    qf[i].q[1] = *(const uint4*)(p + 16);
  }

  v8f o[4];
#pragma unroll
  for (int n = 0; n < 4; ++n) o[n] = zero8();

  float mrow[8], lsum[8];
#pragma unroll
  for (int r = 0; r < 8; ++r) {
    mrow[r] = -3.0e38f;
    lsum[r] = 0.0f;
  }

  const float* mbase = mask + (size_t)b * SS;

  // prologue: stage key block 0 into buffer 0
  async_copy_b128(gK0, &sK[0][kr0 * 64 + ko0]);
  async_copy_b128(gK1, &sK[0][kr1 * 64 + ko1]);
  async_copy_b128(gV0, &sV[0][kr0 * 64 + ko0]);
  async_copy_b128(gV1, &sV[0][kr1 * 64 + ko1]);

  for (int kb = 0, it = 0; kb < SS; kb += 64, ++it) {
    const int cur = it & 1;
    if (kb + 64 < SS) {
      const int nxt = cur ^ 1;
      int kn = kb + 64;
      async_copy_b128(gK0 + kn, &sK[nxt][kr0 * 64 + ko0]);
      async_copy_b128(gK1 + kn, &sK[nxt][kr1 * 64 + ko1]);
      async_copy_b128(gV0 + (size_t)kn * HD, &sV[nxt][kr0 * 64 + ko0]);
      async_copy_b128(gV1 + (size_t)kn * HD, &sV[nxt][kr1 * 64 + ko1]);
      wait_asynccnt<4>();  // oldest 4 (current buffer) complete, in-order
    } else {
      wait_asynccnt<0>();
    }
    __syncthreads();

    const bf16* bK = &sK[cur][0];
    const bf16* bV = &sV[cur][0];

    // scores: four 16x16 tiles (64 keys), inner dim HD=64 via two WMMAs each
    v8f sc[4];
#pragma unroll
    for (int nt = 0; nt < 4; ++nt) {
      Frag b0, b1;
#pragma unroll
      for (int s = 0; s < 16; ++s) {
        int kk = (s & 7) + lhi * 8 + (s >> 3) * 16;
        b0.v[s] = bK[kk * 64 + nt * 16 + lrow];
        b1.v[s] = bK[(32 + kk) * 64 + nt * 16 + lrow];
      }
      v8f tacc = zero8();
      tacc = wmma_bf16(qf[0].v, b0.v, tacc);
      tacc = wmma_bf16(qf[1].v, b1.v, tacc);
      sc[nt] = tacc;
    }

    float mk[4];
#pragma unroll
    for (int nt = 0; nt < 4; ++nt) mk[nt] = mbase[kb + nt * 16 + lrow];

    // online softmax per row (rows live per-VGPR r, replicated over 16 lanes)
    float pr[4][8];
#pragma unroll
    for (int r = 0; r < 8; ++r) {
      float s[4];
#pragma unroll
      for (int nt = 0; nt < 4; ++nt) s[nt] = sc[nt][r] + mk[nt];
      float v = fmaxf(fmaxf(s[0], s[1]), fmaxf(s[2], s[3]));
#pragma unroll
      for (int d = 1; d < 16; d <<= 1) v = fmaxf(v, __shfl_xor(v, d, 32));
      float mnew = fmaxf(mrow[r], v);
      float corr = __expf(mrow[r] - mnew);
      mrow[r] = mnew;
      float rs = 0.0f;
#pragma unroll
      for (int nt = 0; nt < 4; ++nt) {
        float p = __expf(s[nt] - mnew);
        pr[nt][r] = p;
        rs += p;
      }
#pragma unroll
      for (int d = 1; d < 16; d <<= 1) rs += __shfl_xor(rs, d, 32);
      lsum[r] = lsum[r] * corr + rs;
#pragma unroll
      for (int n = 0; n < 4; ++n) o[n][r] *= corr;
    }

    // probs: C layout -> per-wave LDS 16x64 -> reload as two A fragments.
    // DS ops are in-order within a wave; only compiler ordering is needed.
#pragma unroll
    for (int nt = 0; nt < 4; ++nt)
#pragma unroll
      for (int r = 0; r < 8; ++r)
        plds[wid][(r + 8 * lhi) * 64 + nt * 16 + lrow] = (bf16)pr[nt][r];
    asm volatile("" ::: "memory");
    Frag pf[2];
#pragma unroll
    for (int kh = 0; kh < 2; ++kh) {
      const bf16* pp = &plds[wid][lrow * 64 + kh * 32 + lhi * 8];
      pf[kh].q[0] = *(const uint4*)pp;
      pf[kh].q[1] = *(const uint4*)(pp + 16);
    }
    asm volatile("" ::: "memory");

    // O += P * V over 64 keys: two key-halves x four hd tiles
#pragma unroll
    for (int kh = 0; kh < 2; ++kh) {
#pragma unroll
      for (int n = 0; n < 4; ++n) {
        Frag vf;
        const bf16* vp = bV + (kh * 32) * 64 + n * 16 + lrow;
#pragma unroll
        for (int s = 0; s < 16; ++s)
          vf.v[s] = vp[((s & 7) + lhi * 8 + (s >> 3) * 16) * 64];
        o[n] = wmma_bf16(pf[kh].v, vf.v, o[n]);
      }
    }
    __syncthreads();  // all waves done reading before next overwrite
  }

  // normalize and store fp32 ctx [B, S, D]
#pragma unroll
  for (int r = 0; r < 8; ++r) {
    float inv = 1.0f / lsum[r];
    int s = q0 + r + 8 * lhi;
#pragma unroll
    for (int n = 0; n < 4; ++n) {
      out[((size_t)b * SS + s) * DD + h * 64 + n * 16 + lrow] = o[n][r] * inv;
    }
  }
}

// ---------------------------------------------------------------------------
// Host-side launch
// ---------------------------------------------------------------------------
extern "C" void kernel_launch(void* const* d_in, const int* in_sizes, int n_in,
                              void* d_out, int out_size, void* d_ws,
                              size_t ws_size, hipStream_t stream) {
  const float* hs = (const float*)d_in[0];
  const float* mask = (const float*)d_in[1];
  const float* Wq = (const float*)d_in[2];
  const float* bq = (const float*)d_in[3];
  const float* Wk = (const float*)d_in[4];
  const float* bk = (const float*)d_in[5];
  const float* Wv = (const float*)d_in[6];
  const float* bv = (const float*)d_in[7];
  float* out = (float*)d_out;

  char* ws = (char*)d_ws;
  size_t off = 0;
  auto alloc = [&](size_t bytes) -> void* {
    void* p = ws + off;
    off = (off + bytes + 255) & ~(size_t)255;
    return p;
  };

  const size_t nX = (size_t)BB * SS * DD;  // 8M elements
  const size_t nW = (size_t)DD * DD;

  bf16* Xb  = (bf16*)alloc(nX * 2);
  bf16* Wtq = (bf16*)alloc(nW * 2);
  bf16* Wtk = (bf16*)alloc(nW * 2);
  bf16* Wtv = (bf16*)alloc(nW * 2);
  bf16* Qh  = (bf16*)alloc(nX * 2);  // [BH, S, 64]
  bf16* Ktb = (bf16*)alloc(nX * 2);  // [BH, 64, S]
  bf16* Vh  = (bf16*)alloc(nX * 2);  // [BH, S, 64]

  // 1) activations fp32 -> bf16
  f32_to_bf16_kernel<<<(int)((nX + 255) / 256), 256, 0, stream>>>(hs, Xb,
                                                                  (int)nX);
  // 2) weights: transpose + convert
  {
    dim3 tb(16, 16), tg(DD / 16, DD / 16);
    transpose_w_kernel<<<tg, tb, 0, stream>>>(Wq, Wtq);
    transpose_w_kernel<<<tg, tb, 0, stream>>>(Wk, Wtk);
    transpose_w_kernel<<<tg, tb, 0, stream>>>(Wv, Wtv);
  }
  // 3) QKV projections (Q pre-scaled by 1/sqrt(64); K written transposed)
  {
    dim3 gg(DD / 128, (BB * SS) / 128);  // (8, 64)
    qkv_gemm_kernel<<<gg, 128, 0, stream>>>(Xb, Wtq, bq, Qh, 0.125f, 0);
    qkv_gemm_kernel<<<gg, 128, 0, stream>>>(Xb, Wtk, bk, Ktb, 1.0f, 1);
    qkv_gemm_kernel<<<gg, 128, 0, stream>>>(Xb, Wtv, bv, Vh, 1.0f, 0);
  }
  // 4) fused flash attention: 64 (b,h) * 16 blocks, 8 waves x 16 queries each
  attention_kernel<<<BB * HH * (SS / 128), 256, 0, stream>>>(Qh, Ktb, Vh, mask,
                                                             out);
}
